// MultiScaleBlock_2302102470864
// MI455X (gfx1250) — matrix operand order repaired and checked by reading, not compiled, not run on hardware
//
#include <hip/hip_runtime.h>
#include <hip/hip_bf16.h>
#include <math.h>

// ---------------------------------------------------------------------------
// MI455X (gfx1250) MultiScaleBlock.
// ~186 GFLOP vs ~300MB traffic @ 23.3TB/s -> matrix-core bound.
// All heavy math via v_wmma_f32_16x16x32_bf16 (wave32), fragments fed by
// b128 LDS loads from pre-permuted (fragment-ready) tiles. OOB handled by
// index clamping (cheap) instead of zero-fill; all masked results are dead.
// ---------------------------------------------------------------------------

#define BB   4
#define NN   6273      // 1 + 8*28*28
#define CC   768
#define NHh  8
#define HD   96
#define TT   8
#define HH   28
#define WW   28
#define HPp  14
#define WPp  14
#define NQ   1569      // 1 + 8*14*14
#define MX   (BB*NN)   // 25092
#define MQ   (BB*NQ)   // 6276

typedef __attribute__((ext_vector_type(16))) __bf16 v16bf;
typedef __attribute__((ext_vector_type(8)))  float  v8f;

union U16x8  { uint4 q; unsigned short s[8]; };
union FragU  { uint4 q[2]; v16bf v; };

__device__ __forceinline__ unsigned short f2bf(float f) {
  unsigned int u = __float_as_uint(f);
  u += 0x7fffu + ((u >> 16) & 1u);          // round-to-nearest-even
  return (unsigned short)(u >> 16);
}
__device__ __forceinline__ float bf2f(unsigned short h) {
  return __uint_as_float(((unsigned int)h) << 16);
}
__device__ __forceinline__ v8f wmma_bf16(v16bf a, v16bf b, v8f c) {
  return __builtin_amdgcn_wmma_f32_16x16x32_bf16(false, a, false, b, (short)0, c, false, false);
}
// Inverse of the 16-bit A-matrix lane-internal K map (ISA 7.12.2):
// logical k -> position so fragment loads are contiguous. Swaps groups
// 8..15 <-> 16..23 inside each 32-element chunk.
__device__ __forceinline__ int posA(int k) {
  int g = k >> 3;
  return (g == 1) ? k + 8 : (g == 2) ? k - 8 : k;
}
// Load a 32-byte (16 x bf16) lane fragment from 16B-aligned LDS.
__device__ __forceinline__ v16bf lds_frag32(const unsigned short* p) {
  FragU t;
  const uint4* lp = (const uint4*)p;
  t.q[0] = lp[0];
  t.q[1] = lp[1];
  return t.v;
}

// ---------------------------------------------------------------------------
__global__ void cvt_f32_bf16(const float* __restrict__ src,
                             unsigned short* __restrict__ dst, long n) {
  long i = (long)blockIdx.x * blockDim.x + threadIdx.x;
  long stride = (long)gridDim.x * blockDim.x;
  for (; i < n; i += stride) dst[i] = f2bf(src[i]);
}

// LayerNorm over C=768, fp32 in -> bf16 out. One row per block, 256 thr.
__global__ __launch_bounds__(256)
void ln_bf16(const float* __restrict__ x, const float* __restrict__ g,
             const float* __restrict__ b, unsigned short* __restrict__ out,
             int C) {
  int row = blockIdx.x;
  const float* xr = x + (size_t)row * C;
  float s = 0.f, sq = 0.f;
  for (int c = threadIdx.x; c < C; c += blockDim.x) {
    float v = xr[c]; s += v; sq += v * v;
  }
  __shared__ float sb[2][8];
  for (int off = 16; off > 0; off >>= 1) {
    s  += __shfl_down(s,  off, 32);
    sq += __shfl_down(sq, off, 32);
  }
  int wid = threadIdx.x >> 5, lane = threadIdx.x & 31;
  if (lane == 0) { sb[0][wid] = s; sb[1][wid] = sq; }
  __syncthreads();
  if (threadIdx.x == 0) {
    float ts = 0.f, tq = 0.f;
    for (int i = 0; i < 8; i++) { ts += sb[0][i]; tq += sb[1][i]; }
    sb[0][0] = ts; sb[1][0] = tq;
  }
  __syncthreads();
  float mean = sb[0][0] / (float)C;
  float var  = sb[1][0] / (float)C - mean * mean;
  float inv  = rsqrtf(var + 1e-5f);
  unsigned short* orow = out + (size_t)row * C;
  for (int c = threadIdx.x; c < C; c += blockDim.x)
    orow[c] = f2bf((xr[c] - mean) * inv * g[c] + b[c]);
}

// ---------------------------------------------------------------------------
// Generic bf16 WMMA GEMM: out = epi(A[MxK] @ B[KxN] + bias [, res]).
// 128 threads = 4 waves; block tile 64x64; each wave does 16 rows x 64 cols.
// Requires: K % 32 == 0 and N % 64 == 0 (true for all call sites).
// EPI: 0 = bias -> bf16 ; 1 = bias+GELU -> bf16 ; 2 = bias+res(f32) -> f32.
template <int EPI>
__global__ __launch_bounds__(128)
void gemm_bf16(const unsigned short* __restrict__ A,
               const unsigned short* __restrict__ Bw,
               const float* __restrict__ bias,
               const float* __restrict__ res,
               void* __restrict__ outp, int M, int N, int K) {
  __shared__ alignas(16) unsigned short As[64 * 32];  // As[m][k], contiguous k order
  __shared__ alignas(16) unsigned short Bs[64 * 32];  // transposed: Bs[n][k]
  const int bm0 = blockIdx.y * 64, bn0 = blockIdx.x * 64;
  const int tid = threadIdx.x;
  const int wv = tid >> 5, lane = tid & 31;
  const int lrow = lane & 15, lh = lane >> 4;
  v8f acc[4];
  for (int nb = 0; nb < 4; nb++)
    for (int r = 0; r < 8; r++) acc[nb][r] = 0.f;

  // Clamped A row for staging: garbage rows only feed garbage output rows,
  // which the epilogue guard never stores.
  const int a_r = tid >> 1, a_half = tid & 1;
  int a_gm = bm0 + a_r; if (a_gm >= M) a_gm = M - 1;
  const unsigned short* a_base = A + (size_t)a_gm * K + a_half * 16;
  const int b_kk = tid >> 2, b_nseg = tid & 3;
  const unsigned short* b_base = Bw + (size_t)b_kk * N + bn0 + b_nseg * 16;

  for (int k0 = 0; k0 < K; k0 += 32) {
    // Stage A (64x32) in contiguous k order via two 16B stores:
    // half h loads k[h*16..h*16+15] and stores them at pos h*16..h*16+15.
    {
      const uint4* ap = (const uint4*)(a_base + k0);
      uint4 g0 = ap[0], g1 = ap[1];
      *(uint4*)(As + a_r * 32 + a_half * 16)     = g0;
      *(uint4*)(As + a_r * 32 + a_half * 16 + 8) = g1;
    }
    // Stage B transposed: coalesced 32B global reads along N, strided b16
    // stores into Bs[n][k]. Thread t: k = t>>2, 16-wide n segment = t&3.
    {
      U16x8 b0, b1;
      const uint4* bp = (const uint4*)(b_base + (size_t)k0 * N);
      b0.q = bp[0]; b1.q = bp[1];
      for (int i = 0; i < 8; i++) {
        Bs[(b_nseg * 16 + i) * 32 + b_kk]     = b0.s[i];
        Bs[(b_nseg * 16 + 8 + i) * 32 + b_kk] = b1.s[i];
      }
    }
    __syncthreads();
    v16bf a = lds_frag32(As + (wv * 16 + lrow) * 32 + lh * 16);
    for (int nb = 0; nb < 4; nb++) {
      v16bf b = lds_frag32(Bs + (nb * 16 + lrow) * 32 + lh * 16);
      acc[nb] = wmma_bf16(a, b, acc[nb]);
    }
    __syncthreads();
  }
  // C/D layout: VGPR r, lane l -> (M = r + 8*(l>>4), N = l&15)
  for (int nb = 0; nb < 4; nb++) {
    for (int r = 0; r < 8; r++) {
      int m = bm0 + wv * 16 + r + 8 * lh;
      int n = bn0 + nb * 16 + lrow;
      if (m < M) {
        float v = acc[nb][r] + bias[n];
        if (EPI == 1) v = 0.5f * v * (1.f + erff(v * 0.70710678118f));
        if (EPI == 2)
          ((float*)outp)[(size_t)m * N + n] = v + res[(size_t)m * N + n];
        else
          ((unsigned short*)outp)[(size_t)m * N + n] = f2bf(v);
      }
    }
  }
}

// ---------------------------------------------------------------------------
// Depthwise 3x3x3 conv, stride (1,2,2), pad 1, then LN over head-dim (96).
// One block (96 threads) per output token per (which, b, h).
__global__ __launch_bounds__(96)
void pool_conv_ln(const unsigned short* __restrict__ qkv,  // [MX, 2304] bf16
                  const float* __restrict__ wq, const float* __restrict__ wk,
                  const float* __restrict__ wv,
                  const float* __restrict__ gq, const float* __restrict__ bq,
                  const float* __restrict__ gk, const float* __restrict__ bk,
                  const float* __restrict__ gv, const float* __restrict__ bv,
                  unsigned short* __restrict__ pooled) {  // [3][32][NQ][96]
  int idx = blockIdx.x;
  int qi = idx % NQ; int rem = idx / NQ;
  int bh = rem % 32; int which = rem / 32;
  int b = bh >> 3;
  int h = bh & 7;
  int d = threadIdx.x;
  const float* wsel = which == 0 ? wq : (which == 1 ? wk : wv);
  const float* gsel = which == 0 ? gq : (which == 1 ? gk : gv);
  const float* bsel = which == 0 ? bq : (which == 1 ? bk : bv);
  size_t colbase = (size_t)which * CC + (size_t)h * HD + d;
  float val;
  if (qi == 0) {
    val = bf2f(qkv[(size_t)(b * NN) * (3 * CC) + colbase]);
  } else {
    int p = qi - 1;
    int t = p / (HPp * WPp), yz = p % (HPp * WPp);
    int y = yz / WPp, xq = yz % WPp;
    float accv = 0.f;
    for (int dt = 0; dt < 3; dt++) {
      int ti = t + dt - 1; if (ti < 0 || ti >= TT) continue;
      for (int dy = 0; dy < 3; dy++) {
        int yi = 2 * y + dy - 1; if (yi < 0 || yi >= HH) continue;
        for (int dx = 0; dx < 3; dx++) {
          int xi = 2 * xq + dx - 1; if (xi < 0 || xi >= WW) continue;
          int n = 1 + (ti * HH + yi) * WW + xi;
          accv += bf2f(qkv[(size_t)(b * NN + n) * (3 * CC) + colbase]) *
                  wsel[d * 27 + dt * 9 + dy * 3 + dx];
        }
      }
    }
    val = accv;
  }
  __shared__ float red[2][3];
  float s = val, sq = val * val;
  for (int off = 16; off > 0; off >>= 1) {
    s  += __shfl_down(s,  off, 32);
    sq += __shfl_down(sq, off, 32);
  }
  int wid = d >> 5, lane = d & 31;
  if (lane == 0) { red[0][wid] = s; red[1][wid] = sq; }
  __syncthreads();
  float ts = red[0][0] + red[0][1] + red[0][2];
  float tq = red[1][0] + red[1][1] + red[1][2];
  float mean = ts * (1.f / 96.f);
  float var  = tq * (1.f / 96.f) - mean * mean;
  float inv  = rsqrtf(var + 1e-5f);
  float o = (val - mean) * inv * gsel[d] + bsel[d];
  pooled[((size_t)(which * 32 + bh) * NQ + qi) * HD + d] = f2bf(o);
}

// ---------------------------------------------------------------------------
// Flash attention: 1 wave per (b,h,16-query tile); hd=96; keys 32 at a time.
// OOB keys/queries are index-clamped; their contributions are masked (keys)
// or never stored (queries), so no zero-fill is needed.
__global__ __launch_bounds__(32)
void flash_attn(const unsigned short* __restrict__ pooled,
                unsigned short* __restrict__ o_mat) {  // [MQ, 768] bf16
  int qblk = blockIdx.x;
  int bh = blockIdx.y;
  int b = bh >> 3, h = bh & 7;
  int q0 = qblk * 16;
  int lane = threadIdx.x, lrow = lane & 15, lh = lane >> 4;
  const unsigned short* Q  = pooled + (size_t)(0 * 32 + bh) * NQ * HD;
  const unsigned short* Kp = pooled + (size_t)(1 * 32 + bh) * NQ * HD;
  const unsigned short* Vp = pooled + (size_t)(2 * 32 + bh) * NQ * HD;

  __shared__ alignas(16) unsigned short Ks[32 * HD];   // [key][d] row-major
  __shared__ alignas(16) unsigned short Vt[HD * 32];   // [d][key] transposed
  __shared__ alignas(16) unsigned short pT[16 * 32];   // P tile [query][key], contiguous k
  __shared__ float sS[16 * 32];
  __shared__ float corr[16];
  __shared__ float lbuf[16];

  // Q fragments straight from global: per lane-half the A layout is two
  // contiguous 8-element runs -> two 16B loads per 32-chunk.
  v16bf qa[3];
  {
    int qrow = q0 + lrow; if (qrow >= NQ) qrow = NQ - 1;
    const uint4* qbase = (const uint4*)(Q + (size_t)qrow * HD);
    for (int c = 0; c < 3; c++) {
      FragU t;
      t.q[0] = qbase[c * 4 + 2 * lh];      // k = lh?16..23 : 0..7
      t.q[1] = qbase[c * 4 + 2 * lh + 1];  // k = lh?24..31 : 8..15
      qa[c] = t.v;
    }
  }
  v8f oacc[6];
  for (int c = 0; c < 6; c++)
    for (int r = 0; r < 8; r++) oacc[c][r] = 0.f;
  float m_run = -1e30f, l_run = 0.f;   // valid on lanes 0..15
  const float scale = 0.1020620726f;   // 96^-0.5

  for (int kb = 0; kb < NQ; kb += 32) {
    {
      int key = kb + lane; if (key >= NQ) key = NQ - 1;
      const uint4* kp = (const uint4*)(Kp + (size_t)key * HD);
      const uint4* vp = (const uint4*)(Vp + (size_t)key * HD);
      uint4* kd = (uint4*)(Ks + lane * HD);
      for (int i = 0; i < 12; i++) {
        kd[i] = kp[i];
        U16x8 t; t.q = vp[i];            // one 16B chunk live at a time
        for (int jj = 0; jj < 8; jj++)   // transpose V into Vt[d][key]
          Vt[(i * 8 + jj) * 32 + lane] = t.s[jj];
      }
    }
    __syncthreads();
    v8f s0, s1;
    for (int r = 0; r < 8; r++) { s0[r] = 0.f; s1[r] = 0.f; }
    for (int c = 0; c < 3; c++) {
      // S = Q @ K^T: B column n = key row n; B layout contiguous per half.
      v16bf b0 = lds_frag32(Ks + lrow * HD + c * 32 + lh * 16);
      v16bf b1 = lds_frag32(Ks + (16 + lrow) * HD + c * 32 + lh * 16);
      s0 = wmma_bf16(qa[c], b0, s0);
      s1 = wmma_bf16(qa[c], b1, s1);
    }
    {
      bool v0 = (kb + lrow) < NQ, v1 = (kb + 16 + lrow) < NQ;
      for (int r = 0; r < 8; r++) {
        int row = r + 8 * lh;
        sS[row * 32 + lrow]      = v0 ? s0[r] * scale : -1e30f;
        sS[row * 32 + 16 + lrow] = v1 ? s1[r] * scale : -1e30f;
      }
    }
    __syncthreads();
    if (lane < 16) {      // online softmax, one query row per lane
      float mx = -1e30f;
      for (int j = 0; j < 32; j++) mx = fmaxf(mx, sS[lane * 32 + j]);
      float m_new = fmaxf(m_run, mx);
      float cf = __expf(m_run - m_new);
      float ls = l_run * cf;
      for (int j = 0; j < 32; j++) {
        float p = __expf(sS[lane * 32 + j] - m_new);
        ls += p;
        pT[lane * 32 + j] = f2bf(p);  // store in contiguous k order
      }
      corr[lane] = cf;
      m_run = m_new; l_run = ls;
    }
    __syncthreads();
    float cfr[8];
    for (int r = 0; r < 8; r++) cfr[r] = corr[r + 8 * lh];
    for (int c = 0; c < 6; c++)
      for (int r = 0; r < 8; r++) oacc[c][r] *= cfr[r];
    v16bf pa = lds_frag32(pT + lrow * 32 + lh * 16);  // P as A (16x32 keys)
    for (int c = 0; c < 6; c++) {
      // V as B: column = dim d, K = key; Vt[d][key] is contiguous per half.
      v16bf vb = lds_frag32(Vt + (c * 16 + lrow) * 32 + lh * 16);
      oacc[c] = wmma_bf16(pa, vb, oacc[c]);
    }
    __syncthreads();
  }
  if (lane < 16) lbuf[lane] = 1.f / l_run;
  __syncthreads();
  float rden[8];
  for (int r = 0; r < 8; r++) rden[r] = lbuf[r + 8 * lh];
  for (int c = 0; c < 6; c++) {
    for (int r = 0; r < 8; r++) {
      int row = r + 8 * lh;
      int q = q0 + row;
      if (q < NQ) {
        float val = oacc[c][r] * rden[r];
        int d = c * 16 + lrow;
        o_mat[(size_t)(b * NQ + q) * CC + h * HD + d] = f2bf(val);
      }
    }
  }
}

// ---------------------------------------------------------------------------
// Skip path: 3x3 maxpool stride 2 pad 1 over (H,W); cls passthrough. f32 out.
__global__ __launch_bounds__(256)
void pool_skip_k(const float* __restrict__ x, float* __restrict__ skip) {
  int row = blockIdx.x;           // b*NQ + qi
  int b = row / NQ, qi = row % NQ;
  for (int c = threadIdx.x; c < CC; c += 256) {
    float v;
    if (qi == 0) {
      v = x[(size_t)(b * NN) * CC + c];
    } else {
      int p = qi - 1;
      int t = p / (HPp * WPp), yz = p % (HPp * WPp);
      int y = yz / WPp, xq = yz % WPp;
      v = -INFINITY;
      for (int dy = 0; dy < 3; dy++) {
        int yi = 2 * y + dy - 1; if (yi < 0 || yi >= HH) continue;
        for (int dx = 0; dx < 3; dx++) {
          int xi = 2 * xq + dx - 1; if (xi < 0 || xi >= WW) continue;
          int n = 1 + (t * HH + yi) * WW + xi;
          v = fmaxf(v, x[(size_t)(b * NN + n) * CC + c]);
        }
      }
    }
    skip[(size_t)row * CC + c] = v;
  }
}

// ---------------------------------------------------------------------------
extern "C" void kernel_launch(void* const* d_in, const int* in_sizes, int n_in,
                              void* d_out, int out_size, void* d_ws,
                              size_t ws_size, hipStream_t stream) {
  const float* x      = (const float*)d_in[0];
  const float* g1     = (const float*)d_in[1];
  const float* b1     = (const float*)d_in[2];
  const float* W_qkv  = (const float*)d_in[3];
  const float* b_qkv  = (const float*)d_in[4];
  const float* w_pq   = (const float*)d_in[5];
  const float* w_pk   = (const float*)d_in[6];
  const float* w_pv   = (const float*)d_in[7];
  const float* gq     = (const float*)d_in[8];
  const float* bq     = (const float*)d_in[9];
  const float* gk     = (const float*)d_in[10];
  const float* bk     = (const float*)d_in[11];
  const float* gv     = (const float*)d_in[12];
  const float* bv     = (const float*)d_in[13];
  const float* W_proj = (const float*)d_in[14];
  const float* b_proj = (const float*)d_in[15];
  const float* g2     = (const float*)d_in[16];
  const float* b2     = (const float*)d_in[17];
  const float* W_fc1  = (const float*)d_in[18];
  const float* b_fc1  = (const float*)d_in[19];
  const float* W_fc2  = (const float*)d_in[20];
  const float* b_fc2  = (const float*)d_in[21];
  // d_in[22..24] = T,H,W scalars (hardcoded above)

  char* ws = (char*)d_ws;
  size_t off = 0;
  auto carve = [&](size_t bytes) -> void* {
    void* p = ws + off;
    off += (bytes + 255) & ~(size_t)255;
    return p;
  };
  unsigned short* wqkv_bf = (unsigned short*)carve((size_t)CC * 3 * CC * 2);
  unsigned short* wprj_bf = (unsigned short*)carve((size_t)CC * CC * 2);
  unsigned short* wfc1_bf = (unsigned short*)carve((size_t)CC * 4 * CC * 2);
  unsigned short* wfc2_bf = (unsigned short*)carve((size_t)4 * CC * CC * 2);
  unsigned short* xn_bf   = (unsigned short*)carve((size_t)MX * CC * 2);
  unsigned short* qkv_bf  = (unsigned short*)carve((size_t)MX * 3 * CC * 2);
  unsigned short* pooled  = (unsigned short*)carve((size_t)3 * 32 * NQ * HD * 2);
  unsigned short* o_bf    = (unsigned short*)carve((size_t)MQ * CC * 2);
  float*          skip    = (float*)carve((size_t)MQ * CC * 4);
  float*          x2      = (float*)carve((size_t)MQ * CC * 4);
  unsigned short* xn2_bf  = (unsigned short*)carve((size_t)MQ * CC * 2);
  unsigned short* h1_bf   = (unsigned short*)carve((size_t)MQ * 4 * CC * 2);

  // Weight conversions (f32 -> bf16)
  cvt_f32_bf16<<<2048, 256, 0, stream>>>(W_qkv,  wqkv_bf, (long)CC * 3 * CC);
  cvt_f32_bf16<<<1024, 256, 0, stream>>>(W_proj, wprj_bf, (long)CC * CC);
  cvt_f32_bf16<<<2048, 256, 0, stream>>>(W_fc1,  wfc1_bf, (long)CC * 4 * CC);
  cvt_f32_bf16<<<2048, 256, 0, stream>>>(W_fc2,  wfc2_bf, (long)4 * CC * CC);

  // 1) LN(x) -> xn (bf16)
  ln_bf16<<<MX, 256, 0, stream>>>(x, g1, b1, xn_bf, CC);

  // 2) QKV GEMM: [MX,768] @ [768,2304] + bias -> bf16
  gemm_bf16<0><<<dim3((3 * CC + 63) / 64, (MX + 63) / 64), 128, 0, stream>>>(
      xn_bf, wqkv_bf, b_qkv, nullptr, qkv_bf, MX, 3 * CC, CC);

  // 3) Pool conv + LN for q/k/v
  pool_conv_ln<<<3 * 32 * NQ, 96, 0, stream>>>(qkv_bf, w_pq, w_pk, w_pv, gq, bq,
                                               gk, bk, gv, bv, pooled);

  // 4) Attention (flash, WMMA) -> o (bf16, [MQ,768])
  flash_attn<<<dim3((NQ + 15) / 16, BB * NHh), 32, 0, stream>>>(pooled, o_bf);

  // 5) Skip path: maxpool(x)
  pool_skip_k<<<MQ, 256, 0, stream>>>(x, skip);

  // 6) x2 = skip + o @ W_proj + b_proj  (f32)
  gemm_bf16<2><<<dim3((CC + 63) / 64, (MQ + 63) / 64), 128, 0, stream>>>(
      o_bf, wprj_bf, b_proj, skip, x2, MQ, CC, CC);

  // 7) LN(x2) -> xn2 (bf16)
  ln_bf16<<<MQ, 256, 0, stream>>>(x2, g2, b2, xn2_bf, CC);

  // 8) FC1 + exact GELU -> bf16
  gemm_bf16<1><<<dim3((4 * CC + 63) / 64, (MQ + 63) / 64), 128, 0, stream>>>(
      xn2_bf, wfc1_bf, b_fc1, nullptr, h1_bf, MQ, 4 * CC, CC);

  // 9) FC2 + residual x2 -> d_out (f32)
  gemm_bf16<2><<<dim3((CC + 63) / 64, (MQ + 63) / 64), 128, 0, stream>>>(
      h1_bf, wfc2_bf, b_fc2, x2, d_out, MQ, CC, 4 * CC);

  (void)in_sizes; (void)n_in; (void)out_size; (void)ws_size;
}
